// LSTMCell_86449101734018
// MI455X (gfx1250) — matrix operand order repaired and checked by reading, not compiled, not run on hardware
//
#include <hip/hip_runtime.h>
#include <stdint.h>

// ---------- types ----------
typedef __attribute__((ext_vector_type(16))) __bf16 v16bf;
typedef __attribute__((ext_vector_type(8)))  __bf16 v8bf;
typedef __attribute__((ext_vector_type(8)))  float  v8f;
typedef __attribute__((ext_vector_type(4)))  unsigned u32x4;
typedef int v4i __attribute__((vector_size(16)));   // matches builtin's expected pointee

#define Bdim 16384
#define Idim 1024
#define Hdim 1024
#define Kdim 2048       // I + H
#define Ndim 4096       // 4*H
#define LN_EPS 1e-5f

// ---------- async copy (gfx1250 ASYNCcnt path, guarded) ----------
#if __has_builtin(__builtin_amdgcn_global_load_async_to_lds_b128)
#define ASYNC_COPY 1
#endif

__device__ __forceinline__ void async_cp16(const __bf16* g, __bf16* l) {
#if defined(ASYNC_COPY)
  __builtin_amdgcn_global_load_async_to_lds_b128(
      (__attribute__((address_space(1))) v4i*)(uintptr_t)g,
      (__attribute__((address_space(3))) v4i*)(uintptr_t)l,
      0, 0);
#else
  *(u32x4*)l = *(const u32x4*)g;   // global_load_b128 + ds_store_b128
#endif
}

__device__ __forceinline__ void wait_async() {
#if defined(ASYNC_COPY)
#if __has_builtin(__builtin_amdgcn_s_wait_asynccnt)
  __builtin_amdgcn_s_wait_asynccnt(0);
#else
  asm volatile("s_wait_asynccnt 0" ::: "memory");
#endif
#endif
}

// ---------- kernel 1a: build bf16 [x|h] ----------
__global__ void k_cat_bf16(const float* __restrict__ x, const float* __restrict__ h,
                           __bf16* __restrict__ A) {
  size_t idx = (size_t)blockIdx.x * 256 + threadIdx.x;   // Bdim*Kdim threads
  int row = (int)(idx >> 11);
  int col = (int)(idx & 2047);
  float v = (col < Idim) ? x[((size_t)row << 10) + col]
                         : h[((size_t)row << 10) + (col - Idim)];
  A[idx] = (__bf16)v;
}

// ---------- kernel 1b: W -> bf16 ----------
__global__ void k_w_bf16(const float* __restrict__ W, __bf16* __restrict__ Wb) {
  size_t idx = (size_t)blockIdx.x * 256 + threadIdx.x;   // Ndim*Kdim threads
  Wb[idx] = (__bf16)W[idx];
}

// ---------- kernel 2: bf16 WMMA GEMM  T[M,N] = A[M,K] * W[N,K]^T ----------
#define BM 128
#define BN 128
#define BK 64
#define LDT 72            // BK + 8 pad (144B row stride, 16B aligned)

__global__ __launch_bounds__(256)
void k_gemm(const __bf16* __restrict__ A, const __bf16* __restrict__ W,
            float* __restrict__ T) {
  __shared__ __bf16 sA[2][BM * LDT];
  __shared__ __bf16 sB[2][BN * LDT];

  const int tid  = threadIdx.x;
  const int lane = tid & 31;
  const int wave = tid >> 5;          // 0..7
  const int wm   = wave & 3;          // M: wm*32
  const int wn   = wave >> 2;         // N: wn*64
  const int m0   = blockIdx.x * BM;
  const int n0   = blockIdx.y * BN;

  v8f acc[2][4] = {};

  auto stage = [&](int buf, int kt) {
    const __bf16* ga = A + (size_t)m0 * Kdim + (size_t)kt * BK;
    const __bf16* gb = W + (size_t)n0 * Kdim + (size_t)kt * BK;
#pragma unroll
    for (int i = 0; i < 4; ++i) {
      int c    = tid + i * 256;       // 0..1023
      int row  = c >> 3;              // 0..127
      int col8 = (c & 7) * 8;         // element offset, 16B chunks
      async_cp16(ga + (size_t)row * Kdim + col8, &sA[buf][row * LDT + col8]);
      async_cp16(gb + (size_t)row * Kdim + col8, &sB[buf][row * LDT + col8]);
    }
  };

  stage(0, 0);
  wait_async();
  __syncthreads();

  const int half = lane >> 4;   // K-half selector
  const int r    = lane & 15;   // row (A) / column (B) index

  const int nk = Kdim / BK;     // 32
  for (int kt = 0; kt < nk; ++kt) {
    const int cur = kt & 1;
    if (kt + 1 < nk) stage(cur ^ 1, kt + 1);

#pragma unroll
    for (int ks = 0; ks < BK; ks += 32) {
      v16bf af[2], bfr[4];
#pragma unroll
      for (int im = 0; im < 2; ++im) {
        // A 16x32 bf16 layout: lane L holds K {0..7,16..23}; lane L+16 holds {8..15,24..31}
        const __bf16* base = &sA[cur][(wm * 32 + im * 16 + r) * LDT + ks];
        v8bf lo = *(const v8bf*)(base + half * 8);
        v8bf hi = *(const v8bf*)(base + 16 + half * 8);
        af[im] = __builtin_shufflevector(lo, hi, 0,1,2,3,4,5,6,7,8,9,10,11,12,13,14,15);
      }
#pragma unroll
      for (int in = 0; in < 4; ++in) {
        // B 32x16 bf16 layout: lane n holds col n; lanes 0-15 K={0..7,16..23}, lanes 16-31 K={8..15,24..31}
        const __bf16* base = &sB[cur][(wn * 64 + in * 16 + r) * LDT + ks + half * 8];
        v8bf lo = *(const v8bf*)(base);
        v8bf hi = *(const v8bf*)(base + 16);
        bfr[in] = __builtin_shufflevector(lo, hi, 0,1,2,3,4,5,6,7,8,9,10,11,12,13,14,15);
      }
#pragma unroll
      for (int im = 0; im < 2; ++im)
#pragma unroll
        for (int in = 0; in < 4; ++in)
          acc[im][in] = __builtin_amdgcn_wmma_f32_16x16x32_bf16(
              false, af[im], false, bfr[in], (short)0, acc[im][in], false, false);
    }

    wait_async();
    __syncthreads();
  }

  // epilogue: C/D layout — VGPR g: lanes 0-15 -> M=g, lanes 16-31 -> M=8+g, N=lane&15
#pragma unroll
  for (int im = 0; im < 2; ++im)
#pragma unroll
    for (int in = 0; in < 4; ++in) {
      int row0 = m0 + wm * 32 + im * 16 + half * 8;
      int col  = n0 + wn * 64 + in * 16 + r;
#pragma unroll
      for (int g = 0; g < 8; ++g)
        T[(size_t)(row0 + g) * Ndim + col] = acc[im][in][g];
    }
}

// ---------- kernel 3: per-row mean / rstd over 4096 values ----------
__global__ void k_rowstats(const float* __restrict__ T,
                           float* __restrict__ mu, float* __restrict__ rstd) {
  const int row = blockIdx.x;
  const float4* p4 = (const float4*)(T + (size_t)row * Ndim);
  float s = 0.f, s2 = 0.f;
#pragma unroll
  for (int j = 0; j < 4; ++j) {
    float4 v = p4[threadIdx.x + j * 256];
    s  += v.x + v.y + v.z + v.w;
    s2 += v.x * v.x + v.y * v.y + v.z * v.z + v.w * v.w;
  }
#pragma unroll
  for (int off = 16; off > 0; off >>= 1) {
    s  += __shfl_down(s,  off, 32);
    s2 += __shfl_down(s2, off, 32);
  }
  __shared__ float ls[8], ls2[8];
  int lane = threadIdx.x & 31, w = threadIdx.x >> 5;
  if (lane == 0) { ls[w] = s; ls2[w] = s2; }
  __syncthreads();
  if (threadIdx.x == 0) {
    float S = 0.f, S2 = 0.f;
#pragma unroll
    for (int i = 0; i < 8; ++i) { S += ls[i]; S2 += ls2[i]; }
    float m   = S * (1.f / 4096.f);
    float var = S2 * (1.f / 4096.f) - m * m;
    mu[row]   = m;
    rstd[row] = rsqrtf(var + LN_EPS);
  }
}

// ---------- kernel 4: LN + gates + cell update ----------
__global__ void k_gates(const float* __restrict__ T, const float* __restrict__ cell,
                        const float* __restrict__ gamma, const float* __restrict__ beta,
                        const float* __restrict__ mu, const float* __restrict__ rstd,
                        float* __restrict__ out) {
  size_t idx = (size_t)blockIdx.x * 256 + threadIdx.x;  // Bdim*Hdim
  int b = (int)(idx >> 10);
  int h = (int)(idx & 1023);
  const float* tb = T + (size_t)b * Ndim;
  float m = mu[b], r = rstd[b];
  float hd[4];
#pragma unroll
  for (int g = 0; g < 4; ++g)
    hd[g] = (tb[g * Hdim + h] - m) * r * gamma[g * Hdim + h] + beta[g * Hdim + h];
  float fg = 1.f / (1.f + __expf(-hd[0]));
  float ig = 1.f / (1.f + __expf(-hd[1]));
  float og = 1.f / (1.f + __expf(-hd[2]));
  float cand = 0.5f * hd[3] * (1.f + erff(hd[3] * 0.70710678118654752f));
  float nc = fg * cell[idx] + ig * cand;
  out[idx] = og * nc;
  out[(size_t)Bdim * Hdim + idx] = nc;
}

// ---------- launch ----------
extern "C" void kernel_launch(void* const* d_in, const int* in_sizes, int n_in,
                              void* d_out, int out_size, void* d_ws, size_t ws_size,
                              hipStream_t stream) {
  const float* x     = (const float*)d_in[0];
  const float* h     = (const float*)d_in[1];
  const float* cell  = (const float*)d_in[2];
  const float* W     = (const float*)d_in[3];
  const float* gamma = (const float*)d_in[4];
  const float* beta  = (const float*)d_in[5];

  char* ws = (char*)d_ws;
  __bf16* Abf = (__bf16*)ws;                                   // 64 MiB
  __bf16* Wbf = (__bf16*)(ws + (size_t)67108864);              // 16 MiB
  float*  T   = (float*)(ws + (size_t)67108864 + 16777216);    // 256 MiB
  float*  mu  = (float*)(ws + (size_t)352321536);
  float*  rs  = mu + Bdim;

  k_cat_bf16<<<(Bdim * (size_t)Kdim) / 256, 256, 0, stream>>>(x, h, Abf);
  k_w_bf16  <<<(Ndim * (size_t)Kdim) / 256, 256, 0, stream>>>(W, Wbf);

  dim3 grid(Bdim / BM, Ndim / BN);   // 128 x 32
  k_gemm<<<grid, 256, 0, stream>>>(Abf, Wbf, T);

  k_rowstats<<<Bdim, 256, 0, stream>>>(T, mu, rs);

  k_gates<<<(size_t)Bdim * Hdim / 256, 256, 0, stream>>>(T, cell, gamma, beta, mu, rs,
                                                         (float*)d_out);
}
